// GGMLLayer_69930657513655
// MI455X (gfx1250) — compile-verified, matrix operand order checked by reading, not run on hardware
//
#include <hip/hip_runtime.h>
#include <hip/hip_bf16.h>

typedef __attribute__((ext_vector_type(16))) _Float16 v16h;
typedef __attribute__((ext_vector_type(8)))  float    v8f;
typedef __attribute__((ext_vector_type(4)))  float    v4f;
typedef __attribute__((ext_vector_type(4)))  int      v4i;

#define WAVES_PER_BLOCK 4
#define NT 2            // 16-wide N tiles per block
#define THREADS (WAVES_PER_BLOCK * 32)

// Non-temporal weight load: weights are streamed exactly once (180 MB);
// keep x + scales resident in the 192 MB L2 instead.
__device__ __forceinline__ v4i nt4(const int* __restrict__ p) {
    return __builtin_nontemporal_load((const v4i*)p);
}

// Build A fragment (16x32 f16, per-lane layout per CDNA5 ISA 7.12.2):
// lane = m + 16*kgrp ; element i -> K = 16*(i/8) + 8*kgrp + (i%8)
__device__ __forceinline__ v16h load_a_f16(const float* __restrict__ xr, int k0) {
    v4f f0 = *(const v4f*)(xr + k0);
    v4f f1 = *(const v4f*)(xr + k0 + 4);
    v4f f2 = *(const v4f*)(xr + k0 + 16);
    v4f f3 = *(const v4f*)(xr + k0 + 20);
    v16h a;
    #pragma unroll
    for (int i = 0; i < 4; ++i) {
        a[i]      = (_Float16)f0[i];
        a[i + 4]  = (_Float16)f1[i];
        a[i + 8]  = (_Float16)f2[i];
        a[i + 12] = (_Float16)f3[i];
    }
    return a;
}

// Dequantize Q4_0 into a B fragment (32x16 f16): (q-8)*s = fma(q, s, -8s)
// -> v_cvt_f32_i32 + v_fma_mixlo/hi_f16 (no separate add / pk_mul).
__device__ __forceinline__ v16h load_b_q4(const int* __restrict__ wr, int k0, float s) {
    v4i q0 = nt4(wr + k0);
    v4i q1 = nt4(wr + k0 + 4);
    v4i q2 = nt4(wr + k0 + 16);
    v4i q3 = nt4(wr + k0 + 20);
    const float c = -8.0f * s;
    v16h b;
    #pragma unroll
    for (int i = 0; i < 4; ++i) {
        b[i]      = (_Float16)__builtin_fmaf((float)q0[i], s, c);
        b[i + 4]  = (_Float16)__builtin_fmaf((float)q1[i], s, c);
        b[i + 8]  = (_Float16)__builtin_fmaf((float)q2[i], s, c);
        b[i + 12] = (_Float16)__builtin_fmaf((float)q3[i], s, c);
    }
    return b;
}

// One block: NT*16 output columns, all 16 tokens. 4 waves split K; LDS reduce.
__global__ __launch_bounds__(THREADS)
void q4_gemm_wmma(const float* __restrict__ x,     // [16, in_f] fp32
                  const int*   __restrict__ wq,    // [out_f, in_f] int32 nibbles
                  const float* __restrict__ wsc,   // [out_f, in_f/32] fp32
                  const float* __restrict__ bias,  // [out_f]
                  float*       __restrict__ out,   // [16, out_f]
                  int in_f, int out_f)
{
    const int lane = threadIdx.x & 31;
    const int wave = threadIdx.x >> 5;
    const int kgrp = lane >> 4;      // which half of the K-step this lane holds
    const int mn   = lane & 15;      // row (A) / column (B)

    const int nbase = blockIdx.x * (NT * 16);
    const int nblk  = in_f >> 5;                         // scale blocks per row
    const int ksteps_total = in_f >> 5;                  // WMMA K-steps of 32
    const int ksteps_wave  = ksteps_total / WAVES_PER_BLOCK;
    const int ks_begin = wave * ksteps_wave;
    const int ks_end   = ks_begin + ksteps_wave;

    const float* xr  = x   + (size_t)mn * in_f;
    const int*   wr0 = wq  + (size_t)(nbase + 0 * 16 + mn) * in_f;
    const int*   wr1 = wq  + (size_t)(nbase + 1 * 16 + mn) * in_f;
    const float* sc0 = wsc + (size_t)(nbase + 0 * 16 + mn) * nblk;
    const float* sc1 = wsc + (size_t)(nbase + 1 * 16 + mn) * nblk;

    v8f c0 = {};
    v8f c1 = {};

    // 4 K-steps per trip: one float4 scale load per tile, big body to pipeline.
    for (int ks4 = ks_begin; ks4 < ks_end; ks4 += 4) {
        v4f s0 = *(const v4f*)(sc0 + ks4);
        v4f s1 = *(const v4f*)(sc1 + ks4);

        #pragma unroll
        for (int u = 0; u < 4; ++u) {
            const int ks = ks4 + u;
            const int k0 = (ks << 5) + (kgrp << 3);

            // Keep the HBM weight stream ahead of compute.
            __builtin_prefetch(wr0 + k0 + 128, 0, 1);
            __builtin_prefetch(wr1 + k0 + 128, 0, 1);

            v16h a  = load_a_f16(xr, k0);
            v16h b0 = load_b_q4(wr0, k0, s0[u]);
            v16h b1 = load_b_q4(wr1, k0, s1[u]);

            c0 = __builtin_amdgcn_wmma_f32_16x16x32_f16(
                     false, a, false, b0, (short)0, c0, false, false);
            c1 = __builtin_amdgcn_wmma_f32_16x16x32_f16(
                     false, a, false, b1, (short)0, c1, false, false);
        }
    }

    // Deterministic cross-wave K reduction through LDS.
    __shared__ float red[WAVES_PER_BLOCK][NT][256];
    #pragma unroll
    for (int j = 0; j < 8; ++j) {
        red[wave][0][j * 32 + lane] = c0[j];
        red[wave][1][j * 32 + lane] = c1[j];
    }
    __syncthreads();

    if (wave < NT) {
        const int t     = wave;
        const int nglob = nbase + t * 16 + mn;
        const float bv  = bias[nglob];
        #pragma unroll
        for (int j = 0; j < 8; ++j) {
            float s = red[0][t][j * 32 + lane] + red[1][t][j * 32 + lane]
                    + red[2][t][j * 32 + lane] + red[3][t][j * 32 + lane];
            const int m = j + 8 * kgrp;            // C/D layout: M = j + 8*(lane/16)
            out[(size_t)m * out_f + nglob] = s + bv;
        }
    }
}

extern "C" void kernel_launch(void* const* d_in, const int* in_sizes, int n_in,
                              void* d_out, int out_size, void* d_ws, size_t ws_size,
                              hipStream_t stream) {
    const float* x    = (const float*)d_in[0];   // [16, in_f]
    const int*   wq   = (const int*)d_in[1];     // [out_f, in_f]
    const float* wsc  = (const float*)d_in[2];   // [out_f, in_f/32]
    const float* bias = (const float*)d_in[3];   // [out_f]
    float*       out  = (float*)d_out;           // [16, out_f]

    const int out_f = in_sizes[3];
    const int in_f  = in_sizes[1] / out_f;       // 4096
    // out_f = 11008 divisible by NT*16 = 32; in_f = 4096 divisible by 128.
    const int nblocks = out_f / (NT * 16);       // 344

    q4_gemm_wmma<<<nblocks, THREADS, 0, stream>>>(x, wq, wsc, bias, out, in_f, out_f);
}
